// WaveNet_35072702939435
// MI455X (gfx1250) — matrix operand order repaired and checked by reading, not compiled, not run on hardware
//
#include <hip/hip_runtime.h>
#include <hip/hip_bf16.h>

// ---------------------------------------------------------------------------
// CDNA5 (gfx1250) WaveNet forward, bf16 WMMA (v_wmma_f32_16x16x32_bf16).
// h activations stored with a 64-row zero pad per batch so causal conv taps
// never branch:  h[b][64 + t][c],  t in [-64, T).
// ---------------------------------------------------------------------------

typedef __attribute__((ext_vector_type(16))) __bf16 v16bf;
typedef __attribute__((ext_vector_type(8)))  float  v8f;

#define T_LEN 2048
#define PAD   64
#define TP    (T_LEN + PAD)
#define XCS   528               // xc LDS row stride (bf16 elems), padded

union Frag {
    v16bf v;
    uint4 q[2];
    unsigned int u[8];
    unsigned short s[16];
};

__device__ __forceinline__ unsigned short f2bf(float x) {
    unsigned int u = __float_as_uint(x);
    unsigned int r = u + 0x7FFFu + ((u >> 16) & 1u);   // round-to-nearest-even
    return (unsigned short)(r >> 16);
}
__device__ __forceinline__ float bf2f(unsigned short h) {
    return __uint_as_float(((unsigned int)h) << 16);
}

__device__ __forceinline__ v8f wmma_bf16(const Frag& a, const Frag& b, v8f c) {
    return __builtin_amdgcn_wmma_f32_16x16x32_bf16(false, a.v, false, b.v,
                                                   (short)0, c, false, false);
}

// A fragment from packed weights: 32 lanes x 32B, contiguous.
__device__ __forceinline__ Frag load_a(const unsigned int* __restrict__ pw,
                                       int mt, int kc, int KC, int lane) {
    const uint4* p = (const uint4*)(pw + ((size_t)(mt * KC + kc) * 32 + lane) * 8);
    Frag f; f.q[0] = p[0]; f.q[1] = p[1]; return f;
}

// B fragment (K=32 x N=16 bf16) from channel-contiguous storage: 32B per lane.
__device__ __forceinline__ Frag load_b(const unsigned short* __restrict__ h,
                                       long base) {
    const uint4* p = (const uint4*)(h + base);
    Frag f; f.q[0] = p[0]; f.q[1] = p[1]; return f;
}

__device__ __forceinline__ Frag load_b_lds(const unsigned short* g, int off) {
    const uint4* p = (const uint4*)(g + off);
    Frag f; f.q[0] = p[0]; f.q[1] = p[1]; return f;
}

#define ZERO8 {0, 0, 0, 0, 0, 0, 0, 0}

// ---------------------------------------------------------------------------
// Weight packing: source float [M][C][K] -> bf16 A-fragments, col = tap*C + c.
// A-lane layout (16-bit A 16x32):
//   lane l: row m = mt*16 + (l&15)
//   vgpr j, half p:  k = (l>>4)*16 + 2j+p
// ---------------------------------------------------------------------------
__global__ void pack_kernel(const float* __restrict__ src,
                            unsigned int* __restrict__ dst,
                            int C, int K, int KC) {
    int gid  = blockIdx.x * blockDim.x + threadIdx.x;
    int lane = gid & 31;
    int f    = gid >> 5;
    int mt   = f / KC;
    int kc   = f - mt * KC;
    int m    = mt * 16 + (lane & 15);
    int half = lane >> 4;
    unsigned int out[8];
#pragma unroll
    for (int j = 0; j < 8; ++j) {
        unsigned int w = 0;
#pragma unroll
        for (int p = 0; p < 2; ++p) {
            int kl  = 2 * j + p + half * 16;
            int col = kc * 32 + kl;
            int tap = col / C;
            int c   = col - tap * C;
            float v = src[((size_t)m * C + c) * K + tap];
            w |= ((unsigned int)f2bf(v)) << (16 * p);
        }
        out[j] = w;
    }
    unsigned int* d = dst + ((size_t)f * 32 + lane) * 8;
#pragma unroll
    for (int j = 0; j < 8; ++j) d[j] = out[j];
}

// ---------------------------------------------------------------------------
// Zero the 64-row causal pad of a padded activation buffer [B][TP][C].
// ---------------------------------------------------------------------------
__global__ void zero_pad_kernel(unsigned short* __restrict__ buf,
                                int rowElems, int strideElems) {
    int gid = blockIdx.x * 256 + threadIdx.x;
    int b   = gid / rowElems;
    int r   = gid - b * rowElems;
    buf[(size_t)b * strideElems + r] = 0;
}

// ---------------------------------------------------------------------------
// Embedding: h0[b, 64+t, c] = bf16(emb[x[b,t], c])   (E = 256)
// ---------------------------------------------------------------------------
__global__ void embed_kernel(const int* __restrict__ x,
                             const float* __restrict__ emb,
                             unsigned short* __restrict__ h0) {
    int gid = blockIdx.x * 256 + threadIdx.x;
    int c   = gid & 255;
    int bt  = gid >> 8;
    int b   = bt >> 11;            // T = 2048
    int t   = bt & 2047;
    int tok = x[bt];
    h0[((size_t)(b * TP + PAD + t)) * 256 + c] = f2bf(emb[(size_t)tok * 256 + c]);
}

// ---------------------------------------------------------------------------
// Initial causal conv: k=2, d=1, M=128, C=256 (Ktot=512).
// One block = (b, 64-t tile), 8 waves, wave w -> mtile w, 4 N-tiles in regs.
// ---------------------------------------------------------------------------
__global__ void init_conv_kernel(const unsigned short* __restrict__ h0,
                                 const unsigned int* __restrict__ pw,
                                 const float* __restrict__ bias,
                                 unsigned short* __restrict__ hout) {
    const int C = 256, Cout = 128;
    int blk = blockIdx.x;
    int b   = blk >> 5;
    int t0  = (blk & 31) * 64;
    int wave = threadIdx.x >> 5, lane = threadIdx.x & 31;
    int n = lane & 15, half = lane >> 4;
    int mt = wave;
    const int KC = 16;             // 2 taps * 8 chunks of 32
    v8f a0 = ZERO8, a1 = ZERO8, a2 = ZERO8, a3 = ZERO8;
    long rowb = ((long)(b * TP + PAD + t0 + n)) * C + half * 16;
#pragma unroll
    for (int kc = 0; kc < KC; ++kc) {
        int tap = kc >> 3;
        int cchunk = kc & 7;
        long base = rowb + cchunk * 32 - (long)(1 - tap) * C;
        Frag af = load_a(pw, mt, kc, KC, lane);
        Frag b0 = load_b(h0, base);
        Frag b1 = load_b(h0, base + 16 * C);
        Frag b2 = load_b(h0, base + 32 * C);
        Frag b3 = load_b(h0, base + 48 * C);
        a0 = wmma_bf16(af, b0, a0);
        a1 = wmma_bf16(af, b1, a1);
        a2 = wmma_bf16(af, b2, a2);
        a3 = wmma_bf16(af, b3, a3);
    }
    v8f acc[4] = {a0, a1, a2, a3};
#pragma unroll
    for (int nt = 0; nt < 4; ++nt) {
        int t = t0 + nt * 16 + n;
#pragma unroll
        for (int j = 0; j < 8; ++j) {
            int m = mt * 16 + j + half * 8;
            hout[((long)(b * TP + PAD + t)) * Cout + m] = f2bf(acc[nt][j] + bias[m]);
        }
    }
}

// ---------------------------------------------------------------------------
// Fused WaveNet layer: dilated conv (K taps, d=2) -> gate -> res/skip.
// One block = (b, 64-t tile), 256 threads = 8 waves.
// Dynamic LDS: xc[64][XCS] bf16 (t-major) + g[64][256] bf16.
// ---------------------------------------------------------------------------
__global__ void layer_kernel(const unsigned short* __restrict__ hin,
                             unsigned short* __restrict__ hout,
                             float* __restrict__ skips,
                             const unsigned int* __restrict__ pwDil,
                             const unsigned int* __restrict__ pwRes,
                             const unsigned int* __restrict__ pwSkip,
                             const float* __restrict__ dilb,
                             const float* __restrict__ resb,
                             const float* __restrict__ skipb,
                             int K, int first) {
    extern __shared__ unsigned short smem[];
    unsigned short* xc = smem;              // [64][XCS] (t-major, padded)
    unsigned short* g  = smem + 64 * XCS;   // [64][256]
    const int R = 128;
    int blk = blockIdx.x;
    int b   = blk >> 5;
    int t0  = (blk & 31) * 64;
    int wave = threadIdx.x >> 5, lane = threadIdx.x & 31;
    int n = lane & 15, half = lane >> 4;
    int KC = K * 4;                         // K taps * (128/32) chunks
    long rowb = ((long)(b * TP + PAD + t0 + n)) * R + half * 16;

    // -------- Phase A: dilated conv, M=512; 16 live accumulators/wave ------
    {
        v8f acc[4][4];
#pragma unroll
        for (int i = 0; i < 4; ++i)
#pragma unroll
            for (int j = 0; j < 4; ++j) acc[i][j] = (v8f)ZERO8;
        for (int kc = 0; kc < KC; ++kc) {
            int tap = kc >> 2;
            int cchunk = kc & 3;
            long base = rowb + cchunk * 32 - (long)(2 * (K - 1 - tap)) * R;
            Frag b0 = load_b(hin, base);
            Frag b1 = load_b(hin, base + 16 * R);
            Frag b2 = load_b(hin, base + 32 * R);
            Frag b3 = load_b(hin, base + 48 * R);
#pragma unroll
            for (int mt4 = 0; mt4 < 4; ++mt4) {
                Frag af = load_a(pwDil, wave * 4 + mt4, kc, KC, lane);
                acc[mt4][0] = wmma_bf16(af, b0, acc[mt4][0]);
                acc[mt4][1] = wmma_bf16(af, b1, acc[mt4][1]);
                acc[mt4][2] = wmma_bf16(af, b2, acc[mt4][2]);
                acc[mt4][3] = wmma_bf16(af, b3, acc[mt4][3]);
            }
        }
        // epilogue: bias + pack 8 bf16 -> one 16B LDS store per tile
#pragma unroll
        for (int mt4 = 0; mt4 < 4; ++mt4) {
            int mbase = (wave * 4 + mt4) * 16 + half * 8;
#pragma unroll
            for (int nt = 0; nt < 4; ++nt) {
                uint4 st;
                unsigned int pk[4];
#pragma unroll
                for (int jj = 0; jj < 4; ++jj) {
                    float v0 = acc[mt4][nt][2 * jj]     + dilb[mbase + 2 * jj];
                    float v1 = acc[mt4][nt][2 * jj + 1] + dilb[mbase + 2 * jj + 1];
                    pk[jj] = (unsigned int)f2bf(v0) | ((unsigned int)f2bf(v1) << 16);
                }
                st.x = pk[0]; st.y = pk[1]; st.z = pk[2]; st.w = pk[3];
                *(uint4*)(xc + (nt * 16 + n) * XCS + mbase) = st;
            }
        }
    }
    __syncthreads();

    // -------- Phase B: gating  g = filt * sigmoid(gate) --------
    {
        int c0 = (threadIdx.x & 127) * 2;
        int ts = (threadIdx.x >> 7) * 32;
        for (int tt = 0; tt < 32; ++tt) {
            int t  = ts + tt;
            unsigned int uf = *(const unsigned int*)(xc + t * XCS + c0);
            unsigned int ug = *(const unsigned int*)(xc + t * XCS + c0 + 256);
            float f0 = bf2f((unsigned short)(uf & 0xffff));
            float f1 = bf2f((unsigned short)(uf >> 16));
            float s0 = bf2f((unsigned short)(ug & 0xffff));
            float s1 = bf2f((unsigned short)(ug >> 16));
            float g0 = f0 / (1.f + __expf(-s0));
            float g1 = f1 / (1.f + __expf(-s1));
            unsigned int u = (unsigned int)f2bf(g0) | ((unsigned int)f2bf(g1) << 16);
            *(unsigned int*)(g + t * 256 + c0) = u;
        }
    }
    __syncthreads();

    // -------- Phase C1: residual  h' = res_w @ g + res_b + h --------
    {
        int mt = wave;                       // M=128 -> 8 mtiles
        v8f a0 = ZERO8, a1 = ZERO8, a2 = ZERO8, a3 = ZERO8;
#pragma unroll
        for (int kc = 0; kc < 8; ++kc) {
            int co = kc * 32 + half * 16;
            Frag af = load_a(pwRes, mt, kc, 8, lane);
            Frag b0 = load_b_lds(g, (n) * 256 + co);
            Frag b1 = load_b_lds(g, (16 + n) * 256 + co);
            Frag b2 = load_b_lds(g, (32 + n) * 256 + co);
            Frag b3 = load_b_lds(g, (48 + n) * 256 + co);
            a0 = wmma_bf16(af, b0, a0);
            a1 = wmma_bf16(af, b1, a1);
            a2 = wmma_bf16(af, b2, a2);
            a3 = wmma_bf16(af, b3, a3);
        }
        v8f acc[4] = {a0, a1, a2, a3};
#pragma unroll
        for (int nt = 0; nt < 4; ++nt) {
            int t = t0 + nt * 16 + n;
#pragma unroll
            for (int j = 0; j < 8; ++j) {
                int m = mt * 16 + j + half * 8;
                long idx = ((long)(b * TP + PAD + t)) * R + m;
                float hold = bf2f(hin[idx]);
                hout[idx] = f2bf(acc[nt][j] + resb[m] + hold);
            }
        }
    }

    // -------- Phase C2: skip accumulation, M=256; 8 live accumulators ------
    {
        v8f acc[2][4];
#pragma unroll
        for (int i = 0; i < 2; ++i)
#pragma unroll
            for (int j = 0; j < 4; ++j) acc[i][j] = (v8f)ZERO8;
#pragma unroll
        for (int kc = 0; kc < 8; ++kc) {
            int co = kc * 32 + half * 16;
            Frag b0 = load_b_lds(g, (n) * 256 + co);
            Frag b1 = load_b_lds(g, (16 + n) * 256 + co);
            Frag b2 = load_b_lds(g, (32 + n) * 256 + co);
            Frag b3 = load_b_lds(g, (48 + n) * 256 + co);
#pragma unroll
            for (int mh = 0; mh < 2; ++mh) {
                Frag af = load_a(pwSkip, wave * 2 + mh, kc, 8, lane);
                acc[mh][0] = wmma_bf16(af, b0, acc[mh][0]);
                acc[mh][1] = wmma_bf16(af, b1, acc[mh][1]);
                acc[mh][2] = wmma_bf16(af, b2, acc[mh][2]);
                acc[mh][3] = wmma_bf16(af, b3, acc[mh][3]);
            }
        }
#pragma unroll
        for (int mh = 0; mh < 2; ++mh) {
            int mt = wave * 2 + mh;
#pragma unroll
            for (int nt = 0; nt < 4; ++nt) {
                int t = t0 + nt * 16 + n;
#pragma unroll
                for (int j = 0; j < 8; ++j) {
                    int m = mt * 16 + j + half * 8;
                    long idx = ((long)(b * T_LEN + t)) * 256 + m;   // unpadded
                    float v = acc[mh][nt][j] + skipb[m];
                    skips[idx] = first ? v : (skips[idx] + v);
                }
            }
        }
    }
}

// ---------------------------------------------------------------------------
// Output head: out = out_w @ leaky_relu(skips/6) + out_b,  [B,T,V] f32.
// ---------------------------------------------------------------------------
__global__ void output_kernel(const float* __restrict__ skips,
                              const unsigned int* __restrict__ pwOut,
                              const float* __restrict__ outb,
                              float* __restrict__ out) {
    __shared__ unsigned short sb[64 * 256];
    int blk = blockIdx.x;
    int b   = blk >> 5;
    int t0  = (blk & 31) * 64;
    for (int r = 0; r < 64; ++r) {
        float v = skips[((long)(b * T_LEN + t0 + r)) * 256 + threadIdx.x] * (1.f / 6.f);
        v = v > 0.f ? v : 0.01f * v;                    // leaky_relu
        sb[r * 256 + threadIdx.x] = f2bf(v);
    }
    __syncthreads();
    int wave = threadIdx.x >> 5, lane = threadIdx.x & 31;
    int n = lane & 15, half = lane >> 4;
    v8f acc[2][4];
#pragma unroll
    for (int i = 0; i < 2; ++i)
#pragma unroll
        for (int j = 0; j < 4; ++j) acc[i][j] = (v8f)ZERO8;
#pragma unroll
    for (int kc = 0; kc < 8; ++kc) {
        int co = kc * 32 + half * 16;
        Frag b0 = load_b_lds(sb, (n) * 256 + co);
        Frag b1 = load_b_lds(sb, (16 + n) * 256 + co);
        Frag b2 = load_b_lds(sb, (32 + n) * 256 + co);
        Frag b3 = load_b_lds(sb, (48 + n) * 256 + co);
#pragma unroll
        for (int mh = 0; mh < 2; ++mh) {
            Frag af = load_a(pwOut, wave * 2 + mh, kc, 8, lane);
            acc[mh][0] = wmma_bf16(af, b0, acc[mh][0]);
            acc[mh][1] = wmma_bf16(af, b1, acc[mh][1]);
            acc[mh][2] = wmma_bf16(af, b2, acc[mh][2]);
            acc[mh][3] = wmma_bf16(af, b3, acc[mh][3]);
        }
    }
#pragma unroll
    for (int mh = 0; mh < 2; ++mh) {
        int mt = wave * 2 + mh;                         // V=256 -> 16 mtiles
#pragma unroll
        for (int nt = 0; nt < 4; ++nt) {
            int t = t0 + nt * 16 + n;
#pragma unroll
            for (int j = 0; j < 8; ++j) {
                int m = mt * 16 + j + half * 8;
                out[((long)(b * T_LEN + t)) * 256 + m] = acc[mh][nt][j] + outb[m];
            }
        }
    }
}

// ---------------------------------------------------------------------------
// Host launcher.
// d_in order: x, emb, init_w, init_b, dil_w[0..5], dil_b[0..5],
//             res_w[0..5], res_b[0..5], skip_w[0..5], skip_b[0..5],
//             out_w, out_b
// ---------------------------------------------------------------------------
extern "C" void kernel_launch(void* const* d_in, const int* in_sizes, int n_in,
                              void* d_out, int out_size, void* d_ws, size_t ws_size,
                              hipStream_t stream) {
    (void)in_sizes; (void)n_in; (void)out_size; (void)ws_size;
    const int B = 8, NL = 6;
    const int Ks[6] = {1, 2, 4, 8, 16, 32};

    char* ws = (char*)d_ws;
    size_t off = 0;
    auto alloc = [&](size_t bytes) {
        size_t o = off;
        off += (bytes + 255) & ~(size_t)255;
        return o;
    };
    size_t oH0 = alloc((size_t)B * TP * 256 * 2);   // embeddings bf16 (padded)
    size_t oHA = alloc((size_t)B * TP * 128 * 2);   // h ping (padded)
    size_t oHB = alloc((size_t)B * TP * 128 * 2);   // h pong (padded)
    size_t oSK = alloc((size_t)B * T_LEN * 256 * 4);// skips f32
    size_t oPWI = alloc(131072);                    // packed init weights
    size_t oPWD[6], oPWR[6], oPWS[6];
    for (int i = 0; i < NL; ++i) oPWD[i] = alloc((size_t)131072 * Ks[i]);
    for (int i = 0; i < NL; ++i) oPWR[i] = alloc(65536);
    for (int i = 0; i < NL; ++i) oPWS[i] = alloc(131072);
    size_t oPWO = alloc(131072);

    unsigned short* h0 = (unsigned short*)(ws + oH0);
    unsigned short* hA = (unsigned short*)(ws + oHA);
    unsigned short* hB = (unsigned short*)(ws + oHB);
    float*          sk = (float*)(ws + oSK);

    // ---- zero the causal pads (every call: ws is poisoned before timing) ----
    zero_pad_kernel<<<(B * PAD * 256) / 256, 256, 0, stream>>>(h0, PAD * 256, TP * 256);
    zero_pad_kernel<<<(B * PAD * 128) / 256, 256, 0, stream>>>(hA, PAD * 128, TP * 128);
    zero_pad_kernel<<<(B * PAD * 128) / 256, 256, 0, stream>>>(hB, PAD * 128, TP * 128);

    // ---- pack all weights into WMMA A-fragment layout (bf16) ----
    pack_kernel<<<16, 256, 0, stream>>>((const float*)d_in[2],
                                        (unsigned int*)(ws + oPWI), 256, 2, 16);
    for (int i = 0; i < NL; ++i) {
        pack_kernel<<<16 * Ks[i], 256, 0, stream>>>((const float*)d_in[4 + i],
                                                    (unsigned int*)(ws + oPWD[i]),
                                                    128, Ks[i], 4 * Ks[i]);
        pack_kernel<<<8, 256, 0, stream>>>((const float*)d_in[16 + i],
                                           (unsigned int*)(ws + oPWR[i]), 256, 1, 8);
        pack_kernel<<<16, 256, 0, stream>>>((const float*)d_in[28 + i],
                                            (unsigned int*)(ws + oPWS[i]), 256, 1, 8);
    }
    pack_kernel<<<16, 256, 0, stream>>>((const float*)d_in[40],
                                        (unsigned int*)(ws + oPWO), 256, 1, 8);

    // ---- embedding ----
    embed_kernel<<<(B * T_LEN * 256) / 256, 256, 0, stream>>>(
        (const int*)d_in[0], (const float*)d_in[1], h0);

    // ---- init causal conv ----
    init_conv_kernel<<<B * (T_LEN / 64), 256, 0, stream>>>(
        h0, (const unsigned int*)(ws + oPWI), (const float*)d_in[3], hA);

    // ---- 6 fused dilated layers (ping-pong h) ----
    const size_t layer_lds = (64 * XCS + 64 * 256) * sizeof(unsigned short);
    for (int l = 0; l < NL; ++l) {
        unsigned short* hin  = (l & 1) ? hB : hA;
        unsigned short* hout = (l & 1) ? hA : hB;
        layer_kernel<<<B * (T_LEN / 64), 256, layer_lds, stream>>>(
            hin, hout, sk,
            (const unsigned int*)(ws + oPWD[l]),
            (const unsigned int*)(ws + oPWR[l]),
            (const unsigned int*)(ws + oPWS[l]),
            (const float*)d_in[10 + l],
            (const float*)d_in[22 + l],
            (const float*)d_in[34 + l],
            Ks[l], (l == 0) ? 1 : 0);
    }

    // ---- output head ----
    output_kernel<<<B * (T_LEN / 64), 256, 0, stream>>>(
        sk, (const unsigned int*)(ws + oPWO), (const float*)d_in[41],
        (float*)d_out);
}
